// ElementalGTO_86723979641148
// MI455X (gfx1250) — matrix operand order repaired and checked by reading, compiled, not run on hardware
//
#include <hip/hip_runtime.h>

// ---- CDNA5 WMMA types -------------------------------------------------------
typedef __attribute__((ext_vector_type(16))) __bf16       v16bf;
typedef __attribute__((ext_vector_type(8)))  float        v8f;
typedef __attribute__((ext_vector_type(8)))  unsigned int v8u;

#define NATM 128    // atoms per batch (reference: N=128)
#define FPSZ 600    // N_GAUSS(20) * (LMAX+1)(3) * N_MBODY(10)

// float -> bf16 via native conversion (backend emits v_cvt bf16 ops, RNE)
__device__ __forceinline__ __bf16 f2bf(float f) { return (__bf16)f; }

// exchange value with lane^16 (wave32) via ds_swizzle group-of-32 XOR mask 0x10
__device__ __forceinline__ float swz_xor16(float x) {
  union { float f; int i; } a; a.f = x;
  int r = __builtin_amdgcn_ds_swizzle(a.i, 0x401F);
  union { int i; float f; } o; o.i = r;
  return o.f;
}

// One wave32 per (b, n) atom.
// T[c][a][g] accumulated as 4 channels x 2 g-tiles of v_wmma_f32_16x16x32_bf16.
// Species masking of the B tile is a bitwise AND (bf16 zero == all-zero bits).
// All 8 B tiles are materialized before the 8 WMMAs so the matrix pipe streams
// without WAR-hazard NOPs.
__global__ __launch_bounds__(32)
void gto_fp_wmma_kernel(const float* __restrict__ coords,   // (B,128,3)
                        const int*   __restrict__ charges,  // (B,128)
                        const int*   __restrict__ natoms,   // (B,)
                        float*       __restrict__ out)      // (B,128,600)
{
  // LDS staging, per-wave (one wave per workgroup)
  __shared__ __align__(64) __bf16         angT[16][32];  // [a][m] rows 10..15 zero
  __shared__ __align__(64) __bf16         radT[32][32];  // [g][m] rows 20..31 zero
  __shared__ __align__(64) unsigned short mskT[4][32];   // 0xFFFF if species(m)==c

  const int lane = threadIdx.x;           // 0..31
  const int bn   = blockIdx.x;            // b*128 + n
  const int b    = bn >> 7;
  const int n    = bn & (NATM - 1);
  const int na   = natoms[b];

  const __bf16 bf0 = f2bf(0.0f);
  #pragma unroll
  for (int a = 10; a < 16; ++a) angT[a][lane] = bf0;   // pad A rows
  #pragma unroll
  for (int g = 20; g < 32; ++g) radT[g][lane] = bf0;   // pad B cols (g-tile 1)

  const float xn = coords[(size_t)bn * 3 + 0];
  const float yn = coords[(size_t)bn * 3 + 1];
  const float zn = coords[(size_t)bn * 3 + 2];

  v8f acc[8] = {};   // [c*2 + gc]

  #pragma unroll 1
  for (int mc = 0; mc < 4; ++mc) {
    const int m = mc * 32 + lane;
    const size_t mi = (size_t)(b * NATM + m);
    const float mx = coords[mi * 3 + 0];
    const float my = coords[mi * 3 + 1];
    const float mz = coords[mi * 3 + 2];
    const int   zq = charges[mi];

    const float dx = xn - mx, dy = yn - my, dz = zn - mz;
    const float d2 = dx * dx + dy * dy + dz * dz;
    const bool valid = (m != n) && (m < na) && (d2 < 36.0f);
    const float dist = valid ? __builtin_sqrtf(d2) : 1.0f;   // safe placeholder
    const float invd = 1.0f / dist;
    // cosine cutoff * pair-valid mask folded into radial (kills invalid m in B)
    const float cm = valid ? (0.5f * (__cosf(dist * 0.52359877559f) + 1.0f)) : 0.0f;

    // radial[g] = sqrt(2/pi) * exp(-2 (d - 0.3(g+1))^2) * cutoff*mask
    #pragma unroll
    for (int g = 0; g < 20; ++g) {
      const float t = dist - 0.3f * (float)(g + 1);
      radT[g][lane] = f2bf(0.7978845608f * __expf(-2.0f * t * t) * cm);
    }

    // angular: (1/d)^(2+l) * x^p y^q z^r  (l=0:1, l=1:3, l=2:6 comps)
    const float i2 = invd * invd, i3 = i2 * invd, i4 = i2 * i2;
    angT[0][lane] = f2bf(i2);
    angT[1][lane] = f2bf(i3 * dx);
    angT[2][lane] = f2bf(i3 * dy);
    angT[3][lane] = f2bf(i3 * dz);
    angT[4][lane] = f2bf(i4 * dx * dx);
    angT[5][lane] = f2bf(i4 * dx * dy);
    angT[6][lane] = f2bf(i4 * dy * dy);
    angT[7][lane] = f2bf(i4 * dx * dz);
    angT[8][lane] = f2bf(i4 * dy * dz);
    angT[9][lane] = f2bf(i4 * dz * dz);

    // per-channel bit-masks (species(m): 1->0, 6->1, 7->2, 8->3)
    const int sp = (zq == 1) ? 0 : ((zq == 6) ? 1 : ((zq == 7) ? 2 : 3));
    #pragma unroll
    for (int c = 0; c < 4; ++c)
      mskT[c][lane] = (sp == c) ? (unsigned short)0xFFFFu : (unsigned short)0u;
    __syncthreads();

    // --- A matrix (16x32 bf16), contiguous K order, same as B:
    // lane<16: M=lane, K = 0..15; lane>=16: M=lane-16, K = 16..31
    const int aRow = lane & 15;
    const int off8 = (lane & 16) ? 16 : 0;
    v16bf A;
    {
      union { uint4 q[2]; v16bf v; } au;
      au.q[0] = *(const uint4*)&angT[aRow][off8];        // K off8..off8+7
      au.q[1] = *(const uint4*)&angT[aRow][off8 + 8];    // K off8+8..off8+15
      A = au.v;
    }

    // --- channel masks as 8 dwords each (16 bf16 lanes of K)
    const int koff = (lane & 16) ? 16 : 0;
    v8u mk[4];
    #pragma unroll
    for (int c = 0; c < 4; ++c) {
      union { uint4 q[2]; v8u u; } mu;
      mu.q[0] = *(const uint4*)&mskT[c][koff];
      mu.q[1] = *(const uint4*)&mskT[c][koff + 8];
      mk[c] = mu.u;
    }

    // --- B tiles (32x16 bf16): lane<16 holds K=0..15 of column N=lane,
    // lane>=16 holds K=16..31 of column N=lane-16.  Mask via v_and_b32.
    // Build ALL 8 tiles first (distinct registers), then stream 8 WMMAs.
    v16bf Bt[8];
    #pragma unroll
    for (int gc = 0; gc < 2; ++gc) {
      const int g = gc * 16 + (lane & 15);
      v8u rv;
      {
        union { uint4 q[2]; v8u u; } ru;
        ru.q[0] = *(const uint4*)&radT[g][koff];
        ru.q[1] = *(const uint4*)&radT[g][koff + 8];
        rv = ru.u;
      }
      #pragma unroll
      for (int c = 0; c < 4; ++c) {
        union { v8u u; v16bf v; } bu;
        bu.u = rv & mk[c];
        Bt[c * 2 + gc] = bu.v;
      }
    }
    #pragma unroll
    for (int t = 0; t < 8; ++t) {
      acc[t] = __builtin_amdgcn_wmma_f32_16x16x32_bf16(
          false, A, false, Bt[t], (short)0, acc[t], false, false);
    }
    __syncthreads();   // protect LDS before next chunk overwrites
  }

  // ---- epilogue --------------------------------------------------------------
  // C/D layout: lane<16 -> (a = r, g = gc*16+lane); lane>=16 -> (a = r+8).
  // species c:  out[l,c,g]   =   sum_{a in l} w_a T_c[a,g]^2
  // pair (i,j): out[l,4+p,g] = 2*sum_{a in l} w_a T_i[a,g] T_j[a,g]
  // a-group weights: l=0:{a0 w1}; l=1:{a1..a3 w1}; l=2:{a4 1,a5 2,a6 1,a7 2,a8 2,a9 1}
  const float av = (n < na) ? 1.0f : 0.0f;
  const int ci[10] = {0, 1, 2, 3, 0, 0, 0, 1, 1, 2};
  const int cj[10] = {0, 1, 2, 3, 1, 2, 3, 2, 3, 3};
  float* outp = out + (size_t)bn * FPSZ;

  #pragma unroll
  for (int gc = 0; gc < 2; ++gc) {
    const int g = gc * 16 + (lane & 15);
    #pragma unroll
    for (int co = 0; co < 10; ++co) {
      const v8f X = acc[ci[co] * 2 + gc];
      const v8f Y = acc[cj[co] * 2 + gc];
      const float P0 = X[0] * Y[0], P1 = X[1] * Y[1], P2 = X[2] * Y[2], P3 = X[3] * Y[3];
      const float P4 = X[4] * Y[4], P5 = X[5] * Y[5], P6 = X[6] * Y[6], P7 = X[7] * Y[7];
      const float t0 = P0;                                   // l=0 (low lanes)
      const float t1 = P1 + P2 + P3;                         // l=1 (low lanes)
      const float t2 = P4 + 2.0f * P5 + P6 + 2.0f * P7;      // l=2 partial (a4..a7)
      const float h2 = 2.0f * P0 + P1;                       // l=2 partial from a8,a9 (high lanes)
      const float oth = swz_xor16(h2);                       // full wave executes shuffle
      const float f = ((co < 4) ? 1.0f : 2.0f) * av;
      if (lane < 16 && g < 20) {
        outp[      co * 20 + g] = f * t0;
        outp[200 + co * 20 + g] = f * t1;
        outp[400 + co * 20 + g] = f * (t2 + oth);
      }
    }
  }
}

extern "C" void kernel_launch(void* const* d_in, const int* in_sizes, int n_in,
                              void* d_out, int out_size, void* d_ws, size_t ws_size,
                              hipStream_t stream) {
  const float* coords  = (const float*)d_in[0];
  const int*   charges = (const int*)d_in[1];
  const int*   natoms  = (const int*)d_in[2];
  float* out = (float*)d_out;

  const int B = in_sizes[0] / (NATM * 3);   // 32 for the reference setup
  dim3 grid(B * NATM), block(32);
  hipLaunchKernelGGL(gto_fp_wmma_kernel, grid, block, 0, stream,
                     coords, charges, natoms, out);
}
